// GDW_9156870275334
// MI455X (gfx1250) — compile-verified, matrix operand order checked by reading, not run on hardware
//
#include <hip/hip_runtime.h>
#include <hip/hip_bf16.h>
#include <math.h>

using bf16 = __hip_bfloat16;

typedef __bf16 bfraw;
typedef bfraw v16bf __attribute__((ext_vector_type(16)));
typedef bfraw v8bf  __attribute__((ext_vector_type(8)));
typedef float v8f   __attribute__((ext_vector_type(8)));

#define NTHR 256

// ---------------------------------------------------------------------------
// block reductions (256 threads)
// ---------------------------------------------------------------------------
__device__ __forceinline__ float block_reduce_sum(float v, float* sb) {
  int t = threadIdx.x;
  sb[t] = v; __syncthreads();
  for (int o = NTHR / 2; o > 0; o >>= 1) {
    if (t < o) sb[t] += sb[t + o];
    __syncthreads();
  }
  float r = sb[0]; __syncthreads();
  return r;
}
__device__ __forceinline__ float block_reduce_max(float v, float* sb) {
  int t = threadIdx.x;
  sb[t] = v; __syncthreads();
  for (int o = NTHR / 2; o > 0; o >>= 1) {
    if (t < o) sb[t] = fmaxf(sb[t], sb[t + o]);
    __syncthreads();
  }
  float r = sb[0]; __syncthreads();
  return r;
}

// ---------------------------------------------------------------------------
// WMMA bf16 fragment load: 16x32 (or 32x16^T) tile, per ISA 16-bit A layout.
// lane<16 : row r, K {0..7} then {16..23}; lane>=16: row r, K {8..15},{24..31}
// caller folds "half*8" into the pointer; hi chunk is +16 elements.
// ---------------------------------------------------------------------------
__device__ __forceinline__ v16bf load_frag(const bf16* p) {
  v8bf lo = *reinterpret_cast<const v8bf*>(p);
  v8bf hi = *reinterpret_cast<const v8bf*>(p + 16);
  return __builtin_shufflevector(lo, hi, 0, 1, 2, 3, 4, 5, 6, 7,
                                 8, 9, 10, 11, 12, 13, 14, 15);
}

// ---------------------------------------------------------------------------
// Generic GEMM: C[M,N] = A[M,K] @ Bt[N,K]^T, A/Bt row-major bf16, fp32 accum.
// 8 waves (2 along M x 4 along N), wave tile 64x64 (4x4 WMMA tiles),
// block tile 128x256.  Per k-step: 16 b128 loads feed 16 WMMAs (1:1).
// M % 128 == 0, N % 256 == 0, K % 32 == 0 (guaranteed by the problem sizes).
// MODE: 0 = store fp32, 1 = store bf16, 2 = fp32 +=, 3 = exact GELU -> bf16
// ---------------------------------------------------------------------------
template <int MODE>
__launch_bounds__(NTHR)
__global__ void gemm_wmma_k(const bf16* __restrict__ A, const bf16* __restrict__ B,
                            float* __restrict__ C32, bf16* __restrict__ C16,
                            int K, int lda, int ldb, int ldc,
                            long long sA, long long sB, long long sC) {
  int z = blockIdx.z;
  A += (size_t)z * (size_t)sA;
  B += (size_t)z * (size_t)sB;
  size_t coff = (size_t)z * (size_t)sC;

  int lane = threadIdx.x & 31;
  int wave = threadIdx.x >> 5;
  int wm = wave & 1;           // 2 waves along M
  int wn = wave >> 1;          // 4 waves along N
  int m0 = blockIdx.y * 128 + wm * 64;
  int n0 = blockIdx.x * 256 + wn * 64;
  int r = lane & 15;
  int half = lane >> 4;

  const bf16* pa = A + (size_t)(m0 + r) * (size_t)lda + half * 8;
  const bf16* pb = B + (size_t)(n0 + r) * (size_t)ldb + half * 8;

  const v8f vzero = {0.f, 0.f, 0.f, 0.f, 0.f, 0.f, 0.f, 0.f};
  v8f acc[4][4];
#pragma unroll
  for (int i = 0; i < 4; ++i)
#pragma unroll
    for (int j = 0; j < 4; ++j) acc[i][j] = vzero;

  for (int k0 = 0; k0 < K; k0 += 32) {
    v16bf fa[4], fb[4];
#pragma unroll
    for (int i = 0; i < 4; ++i) fa[i] = load_frag(pa + (size_t)(16 * i) * lda + k0);
#pragma unroll
    for (int j = 0; j < 4; ++j) fb[j] = load_frag(pb + (size_t)(16 * j) * ldb + k0);
    __builtin_prefetch(pa + k0 + 256, 0, 0);   // global_prefetch_b8, 8 k-steps ahead
    __builtin_prefetch(pb + k0 + 256, 0, 0);
#pragma unroll
    for (int i = 0; i < 4; ++i)
#pragma unroll
      for (int j = 0; j < 4; ++j)
        acc[i][j] = __builtin_amdgcn_wmma_f32_16x16x32_bf16(
            false, fa[i], false, fb[j], (short)0, acc[i][j], false, false);
  }

#pragma unroll
  for (int i = 0; i < 4; ++i) {
#pragma unroll
    for (int j = 0; j < 4; ++j) {
      int rowb = m0 + 16 * i + 8 * half;  // C layout: VGPR v -> M = v (+8 for lanes 16..31)
      int col = n0 + 16 * j + r;
      v8f a = acc[i][j];
#pragma unroll
      for (int v = 0; v < 8; ++v) {
        size_t idx = coff + (size_t)(rowb + v) * (size_t)ldc + col;
        float val = a[v];
        if constexpr (MODE == 0) {
          C32[idx] = val;
        } else if constexpr (MODE == 1) {
          C16[idx] = __float2bfloat16(val);
        } else if constexpr (MODE == 2) {
          C32[idx] += val;
        } else {
          float g = 0.5f * val * (1.0f + erff(val * 0.70710678118654752f));
          C16[idx] = __float2bfloat16(g);
        }
      }
    }
  }
}

// ---------------------------------------------------------------------------
// Row LayerNorm over D=512 (eps 1e-5, no affine). Optional row gather (idx),
// optional fp32 and bf16 outputs.
// ---------------------------------------------------------------------------
__global__ void k_ln_row(const float* __restrict__ in, const int* __restrict__ idx,
                         float* __restrict__ out32, bf16* __restrict__ out16) {
  __shared__ float sb[NTHR];
  int row = blockIdx.x;
  long long srow = idx ? (long long)idx[row] : (long long)row;
  const float* src = in + (size_t)srow * 512;
  int t = threadIdx.x;
  float x0 = src[t], x1 = src[t + 256];
  float m = block_reduce_sum(x0 + x1, sb) * (1.0f / 512.0f);
  float d0 = x0 - m, d1 = x1 - m;
  float var = block_reduce_sum(d0 * d0 + d1 * d1, sb) * (1.0f / 512.0f);
  float rinv = rsqrtf(var + 1e-5f);
  float y0 = d0 * rinv, y1 = d1 * rinv;
  size_t o = (size_t)row * 512;
  if (out32) { out32[o + t] = y0; out32[o + t + 256] = y1; }
  if (out16) { out16[o + t] = __float2bfloat16(y0); out16[o + t + 256] = __float2bfloat16(y1); }
}

// fp32 -> bf16 elementwise
__global__ void k_convert(const float* __restrict__ in, bf16* __restrict__ out, long long n) {
  long long i = (long long)blockIdx.x * blockDim.x + threadIdx.x;
  long long stride = (long long)gridDim.x * blockDim.x;
  for (; i < n; i += stride) out[i] = __float2bfloat16(in[i]);
}

// batched transpose+convert: in [Z,R,C] fp32 -> out [Z,C,R] bf16
__global__ void k_convT(const float* __restrict__ in, bf16* __restrict__ out,
                        long long R, long long C, long long n) {
  long long i = (long long)blockIdx.x * blockDim.x + threadIdx.x;
  long long stride = (long long)gridDim.x * blockDim.x;
  long long plane = R * C;
  for (; i < n; i += stride) {
    long long z = i / plane, rem = i - z * plane;
    long long r = rem / C, c = rem - r * C;
    out[z * plane + c * R + r] = __float2bfloat16(in[i]);
  }
}

__global__ void k_zero(float* __restrict__ p, long long n) {
  long long i = (long long)blockIdx.x * blockDim.x + threadIdx.x;
  long long stride = (long long)gridDim.x * blockDim.x;
  for (; i < n; i += stride) p[i] = 0.0f;
}

// causal softmax over row s of [Z,S,S] scores; folds 1/sqrt(512) temperature
// and the N_reg = 1/(s+1) row scale into the bf16 attention kernel output.
__global__ void k_softmax_causal(const float* __restrict__ sc, bf16* __restrict__ out, int S) {
  __shared__ float sb[NTHR];
  int s = blockIdx.x, z = blockIdx.y, t = threadIdx.x;
  const float scale = 0.044194173824159216f;  // 1/sqrt(512)
  size_t o = ((size_t)z * S + s) * (size_t)S;
  const float* rowp = sc + o;
  float a0 = (t <= s) ? rowp[t] * scale : -3.4e38f;
  float a1 = (t + 256 <= s) ? rowp[t + 256] * scale : -3.4e38f;
  float mx = block_reduce_max(fmaxf(a0, a1), sb);
  float e0 = (t <= s) ? __expf(a0 - mx) : 0.0f;
  float e1 = (t + 256 <= s) ? __expf(a1 - mx) : 0.0f;
  float sum = block_reduce_sum(e0 + e1, sb);
  float rs = (1.0f / sum) * (1.0f / (float)(s + 1));
  out[o + t] = __float2bfloat16(e0 * rs);
  out[o + t + 256] = __float2bfloat16(e1 * rs);
}

// plain softmax over row v of scores [V,S]; writes transposed Rt[s][v] (bf16)
__global__ void k_softmax_T(const float* __restrict__ sc, bf16* __restrict__ Rt, int V) {
  __shared__ float sb[NTHR];
  int v = blockIdx.x, t = threadIdx.x;
  const float* rowp = sc + (size_t)v * 512;
  float a0 = rowp[t], a1 = rowp[t + 256];
  float mx = block_reduce_max(fmaxf(a0, a1), sb);
  float e0 = __expf(a0 - mx), e1 = __expf(a1 - mx);
  float rs = 1.0f / block_reduce_sum(e0 + e1, sb);
  Rt[(size_t)t * V + v] = __float2bfloat16(e0 * rs);
  Rt[(size_t)(t + 256) * V + v] = __float2bfloat16(e1 * rs);
}

// denom[s] = sum_v Rt[s][v]
__global__ void k_colsum(const bf16* __restrict__ Rt, float* __restrict__ denom, int V) {
  __shared__ float sb[NTHR];
  int s = blockIdx.x;
  float a = 0.0f;
  for (int v = threadIdx.x; v < V; v += NTHR) a += __bfloat162float(Rt[(size_t)s * V + v]);
  float sum = block_reduce_sum(a, sb);
  if (threadIdx.x == 0) denom[s] = sum;
}

// VvT[d][s] = e[s][d] - avg[s][d] / denom[s]   (bf16, transposed for GEMM B)
__global__ void k_vv(const float* __restrict__ e32b, const float* __restrict__ avg,
                     const float* __restrict__ denom, bf16* __restrict__ VvT) {
  int s = blockIdx.x;
  float inv = 1.0f / denom[s];
  for (int d = threadIdx.x; d < 512; d += NTHR) {
    float val = e32b[(size_t)s * 512 + d] - avg[(size_t)s * 512 + d] * inv;
    VvT[(size_t)d * 512 + s] = __float2bfloat16(val);
  }
}

// ---------------------------------------------------------------------------
// host-side GEMM dispatch
// ---------------------------------------------------------------------------
static void launch_gemm(int mode, const bf16* A, const bf16* B, float* C32, bf16* C16,
                        int M, int N, int K, int lda, int ldb, int ldc,
                        long long sA, long long sB, long long sC, int Z, hipStream_t st) {
  dim3 g(N / 256, M / 128, Z), blk(NTHR);
  switch (mode) {
    case 0: gemm_wmma_k<0><<<g, blk, 0, st>>>(A, B, C32, C16, K, lda, ldb, ldc, sA, sB, sC); break;
    case 1: gemm_wmma_k<1><<<g, blk, 0, st>>>(A, B, C32, C16, K, lda, ldb, ldc, sA, sB, sC); break;
    case 2: gemm_wmma_k<2><<<g, blk, 0, st>>>(A, B, C32, C16, K, lda, ldb, ldc, sA, sB, sC); break;
    default: gemm_wmma_k<3><<<g, blk, 0, st>>>(A, B, C32, C16, K, lda, ldb, ldc, sA, sB, sC); break;
  }
}

extern "C" void kernel_launch(void* const* d_in, const int* in_sizes, int n_in,
                              void* d_out, int out_size, void* d_ws, size_t ws_size,
                              hipStream_t stream) {
  (void)in_sizes; (void)n_in; (void)out_size; (void)ws_size;
  const int B = 2, S = 512, V = 32000, D = 512, H = 8, L = 4;
  const int HD = H * D, F4 = 4 * D;

  const int*   x    = (const int*)d_in[0];
  const float* wte  = (const float*)d_in[1];
  const float* wpe  = (const float*)d_in[2];
  const float* pWqk = (const float*)d_in[3];
  const float* eWqk = (const float*)d_in[4];
  const float* pWo  = (const float*)d_in[5];
  const float* eWo  = (const float*)d_in[6];
  const float* ffw1 = (const float*)d_in[7];
  const float* ffw2 = (const float*)d_in[8];
  float* out = (float*)d_out;

  // deterministic bump allocator on d_ws
  char* base = (char*)d_ws;
  size_t off = 0;
  auto alloc = [&](size_t bytes) -> void* {
    void* p = base + off;
    off = (off + bytes + 255) & ~(size_t)255;
    return p;
  };

  float* e32     = (float*)alloc((size_t)B * S * D * 4);
  bf16*  e16     = (bf16*)alloc((size_t)B * S * D * 2);
  bf16*  p16     = (bf16*)alloc((size_t)(S + 1) * D * 2);
  bf16*  wte16   = (bf16*)alloc((size_t)V * D * 2);
  bf16*  wteT16  = (bf16*)alloc((size_t)V * D * 2);
  bf16*  wteln16 = (bf16*)alloc((size_t)V * D * 2);
  bf16*  pWqkT16 = (bf16*)alloc((size_t)H * D * D * 2);
  bf16*  eWqkT16 = (bf16*)alloc((size_t)H * D * D * 2);
  bf16*  pWo16   = (bf16*)alloc((size_t)L * D * HD * 2);
  bf16*  eWo16   = (bf16*)alloc((size_t)L * D * HD * 2);
  bf16*  ffw116  = (bf16*)alloc((size_t)F4 * D * 2);
  bf16*  ffw216  = (bf16*)alloc((size_t)D * F4 * 2);
  bf16*  pQ16    = (bf16*)alloc((size_t)H * S * D * 2);
  bf16*  pK16    = (bf16*)alloc((size_t)H * S * D * 2);
  bf16*  eQK16   = (bf16*)alloc((size_t)B * H * S * D * 2);
  bf16*  pkrn16  = (bf16*)alloc((size_t)H * S * S * 2);
  bf16*  ekrn16  = (bf16*)alloc((size_t)B * H * S * S * 2);
  float* scores  = (float*)alloc((size_t)V * S * 4);  // also holds [B*H,S,S] (smaller)
  bf16*  Rt16    = (bf16*)alloc((size_t)S * V * 2);
  float* denom   = (float*)alloc((size_t)S * 4);
  float* avg32   = (float*)alloc((size_t)S * D * 4);
  bf16*  VvT16   = (bf16*)alloc((size_t)B * D * S * 2);
  bf16*  pdc16   = (bf16*)alloc((size_t)B * S * HD * 2);
  bf16*  edc16   = (bf16*)alloc((size_t)B * S * HD * 2);
  float* f32     = (float*)alloc((size_t)B * S * D * 4);
  bf16*  f16     = (bf16*)alloc((size_t)B * S * D * 2);
  bf16*  g16     = (bf16*)alloc((size_t)B * S * D * 2);
  bf16*  h116    = (bf16*)alloc((size_t)B * S * F4 * 2);

  const long long llD = D, llS = S, llV = V;

  // ---- phase 0: LN + precision conversions ----
  k_ln_row<<<B * S, NTHR, 0, stream>>>(wte, x, e32, e16);                  // e = LN(wte[x])
  k_ln_row<<<S + 1, NTHR, 0, stream>>>(wpe, nullptr, nullptr, p16);        // p = LN(wpe[:S+1])
  k_ln_row<<<V, NTHR, 0, stream>>>(wte, nullptr, nullptr, wteln16);        // LN(wte)
  k_convert<<<4096, NTHR, 0, stream>>>(wte, wte16, (long long)V * D);
  k_convT<<<4096, NTHR, 0, stream>>>(wte, wteT16, llV, llD, (long long)V * D);
  k_convT<<<4096, NTHR, 0, stream>>>(pWqk, pWqkT16, llD, llD, (long long)H * D * D);
  k_convT<<<4096, NTHR, 0, stream>>>(eWqk, eWqkT16, llD, llD, (long long)H * D * D);
  k_convert<<<4096, NTHR, 0, stream>>>(pWo, pWo16, (long long)L * D * HD);
  k_convert<<<4096, NTHR, 0, stream>>>(eWo, eWo16, (long long)L * D * HD);
  k_convert<<<2048, NTHR, 0, stream>>>(ffw1, ffw116, (long long)F4 * D);
  k_convert<<<2048, NTHR, 0, stream>>>(ffw2, ffw216, (long long)D * F4);
  k_zero<<<2048, NTHR, 0, stream>>>(f32, (long long)B * S * D);

  // ---- phase 1: attention kernels (position + embedding) ----
  launch_gemm(1, p16, pWqkT16, nullptr, pK16, S, D, D, D, D, D, 0, llD * D, llS * D, H, stream);
  launch_gemm(1, p16 + D, pWqkT16, nullptr, pQ16, S, D, D, D, D, D, 0, llD * D, llS * D, H, stream);
  launch_gemm(0, pQ16, pK16, scores, nullptr, S, S, D, D, D, S, llS * D, llS * D, llS * S, H, stream);
  k_softmax_causal<<<dim3(S, H), NTHR, 0, stream>>>(scores, pkrn16, S);
  for (int b = 0; b < B; ++b)
    launch_gemm(1, e16 + (size_t)b * S * D, eWqkT16, nullptr, eQK16 + (size_t)b * H * S * D,
                S, D, D, D, D, D, 0, llD * D, llS * D, H, stream);
  launch_gemm(0, eQK16, eQK16, scores, nullptr, S, S, D, D, D, S, llS * D, llS * D, llS * S, B * H, stream);
  k_softmax_causal<<<dim3(S, B * H), NTHR, 0, stream>>>(scores, ekrn16, S);

  // ---- phase 2: L gd-steps ----
  for (int k = 0; k < L; ++k) {
    k_convert<<<2048, NTHR, 0, stream>>>(f32, f16, (long long)B * S * D);
    for (int b = 0; b < B; ++b) {
      // scores[v,s] = wte @ f[b]^T
      launch_gemm(0, wte16, f16 + (size_t)b * S * D, scores, nullptr,
                  V, S, D, D, D, S, 0, 0, 0, 1, stream);
      k_softmax_T<<<V, NTHR, 0, stream>>>(scores, Rt16, V);          // softmax over s, write R^T
      k_colsum<<<S, NTHR, 0, stream>>>(Rt16, denom, V);              // sum over v
      launch_gemm(0, Rt16, wteT16, avg32, nullptr, S, D, V, V, V, D, 0, 0, 0, 1, stream);
      k_vv<<<S, NTHR, 0, stream>>>(e32 + (size_t)b * S * D, avg32, denom,
                                   VvT16 + (size_t)b * D * S);
    }
    for (int b = 0; b < B; ++b) {
      // pd/ed concat [S, H*D]: A = kernel[h] (S,S), Bt = Vv^T (D,S); N_reg already folded in
      launch_gemm(1, pkrn16, VvT16 + (size_t)b * D * S, nullptr, pdc16 + (size_t)b * S * HD,
                  S, D, S, S, S, HD, llS * S, 0, llD, H, stream);
      launch_gemm(1, ekrn16 + (size_t)b * H * S * S, VvT16 + (size_t)b * D * S, nullptr,
                  edc16 + (size_t)b * S * HD, S, D, S, S, S, HD, llS * S, 0, llD, H, stream);
    }
    // f += pd @ p_W_o[k]^T ; f += ed @ e_W_o[k]^T
    launch_gemm(2, pdc16, pWo16 + (size_t)k * D * HD, f32, nullptr,
                B * S, D, HD, HD, HD, D, 0, 0, 0, 1, stream);
    launch_gemm(2, edc16, eWo16 + (size_t)k * D * HD, f32, nullptr,
                B * S, D, HD, HD, HD, D, 0, 0, 0, 1, stream);
  }

  // ---- phase 3: feed-forward ----
  k_ln_row<<<B * S, NTHR, 0, stream>>>(f32, nullptr, nullptr, g16);
  launch_gemm(3, g16, ffw116, nullptr, h116, B * S, F4, D, D, D, F4, 0, 0, 0, 1, stream);  // GELU
  launch_gemm(2, h116, ffw216, f32, nullptr, B * S, D, F4, F4, F4, D, 0, 0, 0, 1, stream); // f += ...

  // ---- phase 4: logits = f @ LN(wte)^T ----
  k_convert<<<2048, NTHR, 0, stream>>>(f32, f16, (long long)B * S * D);
  launch_gemm(0, f16, wteln16, out, nullptr, B * S, V, D, D, D, V, 0, 0, 0, 1, stream);
}
